// TransformerBlock_24292335027129
// MI455X (gfx1250) — compile-verified, matrix operand order checked by reading, not run on hardware
//
#include <hip/hip_runtime.h>
#include <hip/hip_bf16.h>
#include <math.h>

// ---------------------------------------------------------------------------
// Types
// ---------------------------------------------------------------------------
typedef __bf16 bf16_t;
typedef __attribute__((ext_vector_type(16))) __bf16 v16bf;
typedef __attribute__((ext_vector_type(8)))  __bf16 v8bf;
typedef __attribute__((ext_vector_type(8)))  float  v8f;
typedef __attribute__((ext_vector_type(4)))  unsigned int u32x4;
typedef __attribute__((ext_vector_type(8)))  int          i32x8;
typedef __attribute__((ext_vector_type(4)))  int          i32x4;

#define SEQ     4096
#define EMBED   768
#define FFN_DIM 3072
#define NHEADS  12
#define HDIM    64

static __device__ __forceinline__ v8f vzero8() {
  v8f z;
#pragma unroll
  for (int i = 0; i < 8; ++i) z[i] = 0.0f;
  return z;
}

// 16-bit A/B fragment loader (16x32 tile, row-major, stride `ld` elems).
// Lane layout per CDNA5 ISA 7.12.2: row = lane&15; lanes<16 hold K=kb..kb+7
// and kb+16..kb+23 with kb = (lane>>4)*8. Two 16B loads -> ds_load_b128 x2.
static __device__ __forceinline__ v16bf load_frag(const bf16_t* __restrict__ base,
                                                  int ld, int lane) {
  const int r  = lane & 15;
  const int kb = (lane >> 4) << 3;
  const bf16_t* p = base + r * ld + kb;
  v8bf lo = *(const v8bf*)(p);
  v8bf hi = *(const v8bf*)(p + 16);
  v16bf f;
#pragma unroll
  for (int i = 0; i < 8; ++i) { f[i] = lo[i]; f[i + 8] = hi[i]; }
  return f;
}

static __device__ __forceinline__ v8f wmma_bf16(v16bf a, v16bf b, v8f c) {
  // (neg_a, A, neg_b, B, c_mod, C, reuse_a, reuse_b)
  return __builtin_amdgcn_wmma_f32_16x16x32_bf16(false, a, false, b, (short)0, c,
                                                 false, false);
}

// ---------------------------------------------------------------------------
// TDM: DMA a 128x32 bf16 tile (row-major, global stride K elems) into LDS at
// byte offset lds_off, with HW padding 4 dwords per 16 dwords -> 40-elem rows.
// D# layout per CDNA5 ISA ch.8 (CDNA5 ISA). This toolchain's builtin
// is the 6-arg form: (g0 u32x4, g1 i32x8, g2 i32x4, g3 i32x4, i32x8, cpol).
// ---------------------------------------------------------------------------
static __device__ __forceinline__ void tdm_load_tile_a(const bf16_t* gaddr,
                                                       unsigned lds_off,
                                                       int K, int M) {
  const unsigned long long ga = (unsigned long long)(size_t)gaddr;
  union { unsigned w[4]; u32x4 v; } g0;
  g0.w[0] = 1u;                                       // count=1 (valid), user mode
  g0.w[1] = lds_off;                                  // lds_addr
  g0.w[2] = (unsigned)(ga & 0xFFFFFFFFu);             // global_addr[31:0]
  g0.w[3] = (unsigned)((ga >> 32) & 0x1FFFFFFu)       // global_addr[56:32]
            | (2u << 30);                             // type=2 ("image")
  union { unsigned w[8]; i32x8 v; } g1;
  g1.w[0] = (1u << 16)                                // data_size = 1 (2 bytes)
          | (1u << 20)                                // pad_enable
          | (3u << 22)                                // pad_interval: 16 dwords
          | (3u << 25);                               // pad_amount:   4 dwords
  g1.w[1] = ((unsigned)K & 0xFFFFu) << 16;            // tensor_dim0[15:0] @bit48
  g1.w[2] = ((unsigned)K >> 16) | (((unsigned)M & 0xFFFFu) << 16); // dim0 hi, dim1 lo
  g1.w[3] = ((unsigned)M >> 16) | (32u << 16);        // dim1 hi, tile_dim0=32
  g1.w[4] = 128u;                                     // tile_dim1=128, tile_dim2=0
  g1.w[5] = (unsigned)K;                              // tensor_dim0_stride[31:0]
  g1.w[6] = 0u;                                       // stride hi, dim1_stride lo
  g1.w[7] = 0u;
  const i32x4 z4 = {0, 0, 0, 0};                      // groups 2/3: 2D tensor
  const i32x8 z8 = {0, 0, 0, 0, 0, 0, 0, 0};
  __builtin_amdgcn_tensor_load_to_lds(g0.v, (i32x8)g1.v, z4, z4, z8, 0);
}

// ---------------------------------------------------------------------------
// f32 -> bf16 conversion
// ---------------------------------------------------------------------------
__global__ void cvt_f32_bf16(const float* __restrict__ in, bf16_t* __restrict__ out,
                             int n) {
  int i = blockIdx.x * 256 + threadIdx.x;
  if (i < n) out[i] = (bf16_t)in[i];
}

// ---------------------------------------------------------------------------
// Tiled bf16 WMMA GEMM: C[M,N] = A[M,K] @ B[K,N] + bias
// Block: 128x128 tile, 256 threads (8 waves as 2x4), BK=32, double-buffered
// dynamic LDS. A tiles staged by the Tensor Data Mover (wave 0, TENSORcnt);
// B tiles staged manually (needs transpose, which TDM cannot do).
// LDS layout (elements): A buf0 @0, A buf1 @5120, B buf0 @10240, B buf1 @15360
// EPI: 0 = store f32; 2 = GELU -> bf16; 3 = bf16, head-major [H][M][64]
// ---------------------------------------------------------------------------
template <int EPI>
__global__ __launch_bounds__(256) void gemm_bf16_wmma(
    const bf16_t* __restrict__ A, const bf16_t* __restrict__ B,
    const float* __restrict__ bias,
    float* __restrict__ outF, bf16_t* __restrict__ outB,
    int M, int N, int K) {
  extern __shared__ bf16_t smem[];

  const int tid  = threadIdx.x;
  const int lane = tid & 31;
  const int wave = tid >> 5;
  const int m0   = blockIdx.y * 128;
  const int n0   = blockIdx.x * 128;
  const int wm   = (wave >> 2) * 64;   // wave row offset (0/64)
  const int wn   = (wave & 3) * 32;    // wave col offset (0..96)

  const int rB = tid >> 3, cB = (tid & 7) * 16;   // 32 rows x 128 cols

  auto stageA = [&](int buf, int k0) {             // TDM, one wave per block
    if (wave == 0)
      tdm_load_tile_a(A + (size_t)m0 * K + k0, (unsigned)(buf * 10240), K, M);
  };
  auto stageB = [&](int buf, int k0) {             // transpose scatter to [n][k]
    bf16_t* dst = smem + 10240 + buf * 5120;
    const bf16_t* gb = B + (size_t)(k0 + rB) * N + n0 + cB;
    v8bf b0 = *(const v8bf*)gb;
    v8bf b1 = *(const v8bf*)(gb + 8);
#pragma unroll
    for (int i = 0; i < 8; ++i) dst[(cB + i) * 40 + rB] = b0[i];
#pragma unroll
    for (int i = 0; i < 8; ++i) dst[(cB + 8 + i) * 40 + rB] = b1[i];
  };

  v8f acc[4][2];
#pragma unroll
  for (int i = 0; i < 4; ++i)
#pragma unroll
    for (int j = 0; j < 2; ++j) acc[i][j] = vzero8();

  stageA(0, 0);
  stageB(0, 0);
  if (wave == 0) __builtin_amdgcn_s_wait_tensorcnt((short)0);
  __syncthreads();
  const int KT = K >> 5;
  for (int kt = 0; kt < KT; ++kt) {
    const int cur = kt & 1;
    if (kt + 1 < KT) {
      stageA(cur ^ 1, (kt + 1) << 5);
      stageB(cur ^ 1, (kt + 1) << 5);
    }
    const bf16_t* sAc = smem + cur * 5120;
    const bf16_t* sBc = smem + 10240 + cur * 5120;
    v16bf af[4], bfr[2];
#pragma unroll
    for (int i = 0; i < 4; ++i)
      af[i] = load_frag(sAc + (wm + i * 16) * 40, 40, lane);
#pragma unroll
    for (int j = 0; j < 2; ++j)
      bfr[j] = load_frag(sBc + (wn + j * 16) * 40, 40, lane);
#pragma unroll
    for (int i = 0; i < 4; ++i)
#pragma unroll
      for (int j = 0; j < 2; ++j) acc[i][j] = wmma_bf16(af[i], bfr[j], acc[i][j]);
    if (wave == 0) __builtin_amdgcn_s_wait_tensorcnt((short)0);
    __syncthreads();
  }

  // Epilogue. C layout: VGPR e -> row (lane>>4)*8+e, col = lane&15 (ISA 7.12.2)
  const int rb = (lane >> 4) << 3;
  const int cl = lane & 15;
#pragma unroll
  for (int j = 0; j < 2; ++j) {
    const int n = n0 + wn + j * 16 + cl;
    const float bv = bias[n];
#pragma unroll
    for (int i = 0; i < 4; ++i) {
#pragma unroll
      for (int e = 0; e < 8; ++e) {
        const int m = m0 + wm + i * 16 + rb + e;
        float v = acc[i][j][e] + bv;
        if constexpr (EPI == 0) {
          outF[(size_t)m * N + n] = v;
        } else if constexpr (EPI == 2) {
          v = 0.5f * v * (1.0f + erff(v * 0.70710678118f));   // exact GELU
          outB[(size_t)m * N + n] = (bf16_t)v;
        } else if constexpr (EPI == 3) {
          const int h = n >> 6, d = n & 63;                    // [H][M][64]
          outB[(((size_t)h * M + m) << 6) + d] = (bf16_t)v;
        }
      }
    }
  }
}

// ---------------------------------------------------------------------------
// Flash attention: Q,K,V in [H][S][64] bf16 -> ctx [S][768] bf16
// Block = (head, 128 queries); each wave owns 16 query rows; key chunks of 64.
// ---------------------------------------------------------------------------
__global__ __launch_bounds__(256) void flash_attn_wmma(
    const bf16_t* __restrict__ Q, const bf16_t* __restrict__ Kx,
    const bf16_t* __restrict__ V, bf16_t* __restrict__ ctx, int S) {
  __shared__ bf16_t sK[64 * 72];      // key chunk, row-major [key][dim]
  __shared__ bf16_t sVt[64 * 72];     // V transposed [dim][key]
  __shared__ bf16_t sP[8][16 * 72];   // per-wave probabilities [row][key]

  const int tid = threadIdx.x, lane = tid & 31, wave = tid >> 5;
  const int h = blockIdx.y;
  const size_t headOff = (size_t)h * S * HDIM;
  const int q0 = blockIdx.x * 128 + wave * 16;
  const int rb = (lane >> 4) << 3;
  const int cl = lane & 15;

  // Q fragments: 16 rows x 64 dims = 2 K-steps, straight from global memory.
  const v16bf qf0 = load_frag(Q + headOff + (size_t)q0 * HDIM, HDIM, lane);
  const v16bf qf1 = load_frag(Q + headOff + (size_t)q0 * HDIM + 32, HDIM, lane);

  v8f o[4];
#pragma unroll
  for (int dt = 0; dt < 4; ++dt) o[dt] = vzero8();
  float rmax[8], rsum[8];
#pragma unroll
  for (int e = 0; e < 8; ++e) { rmax[e] = -1e30f; rsum[e] = 0.0f; }

  const float SCL = 0.125f * 1.4426950408889634f;   // (1/sqrt(64)) * log2(e)
  const int r = tid >> 2, c = (tid & 3) * 16;       // 64 x 64 chunk staging

  for (int k0 = 0; k0 < S; k0 += 64) {
    {   // cooperative stage of K (row-major) and V (transposed)
      const bf16_t* gk = Kx + headOff + (size_t)(k0 + r) * HDIM + c;
      v8bf ka = *(const v8bf*)gk;
      v8bf kb = *(const v8bf*)(gk + 8);
      *(v8bf*)&sK[r * 72 + c]     = ka;
      *(v8bf*)&sK[r * 72 + c + 8] = kb;
      const bf16_t* gv = V + headOff + (size_t)(k0 + r) * HDIM + c;
      v8bf va = *(const v8bf*)gv;
      v8bf vb = *(const v8bf*)(gv + 8);
#pragma unroll
      for (int i = 0; i < 8; ++i) sVt[(c + i) * 72 + r] = va[i];
#pragma unroll
      for (int i = 0; i < 8; ++i) sVt[(c + 8 + i) * 72 + r] = vb[i];
    }
    __syncthreads();

    // scores = Q @ K^T : 4 key tiles of 16, k = 64 -> two WMMAs each
    v8f sc[4];
#pragma unroll
    for (int nt = 0; nt < 4; ++nt) {
      v8f a = vzero8();
      a = wmma_bf16(qf0, load_frag(&sK[(nt * 16) * 72], 72, lane), a);
      a = wmma_bf16(qf1, load_frag(&sK[(nt * 16) * 72 + 32], 72, lane), a);
      sc[nt] = a;
    }

    // online softmax per query row (rows live in VGPR index e, 16-lane halves)
#pragma unroll
    for (int e = 0; e < 8; ++e) {
      float mx = sc[0][e];
#pragma unroll
      for (int nt = 1; nt < 4; ++nt) mx = fmaxf(mx, sc[nt][e]);
#pragma unroll
      for (int msk = 8; msk >= 1; msk >>= 1) mx = fmaxf(mx, __shfl_xor(mx, msk));
      const float nm = fmaxf(rmax[e], mx);
      const float alpha = exp2f((rmax[e] - nm) * SCL);
      rmax[e] = nm;
      float ps = 0.0f;
#pragma unroll
      for (int nt = 0; nt < 4; ++nt) {
        const float p = exp2f((sc[nt][e] - nm) * SCL);
        sc[nt][e] = p;
        ps += p;
      }
#pragma unroll
      for (int msk = 8; msk >= 1; msk >>= 1) ps += __shfl_xor(ps, msk);
      rsum[e] = rsum[e] * alpha + ps;
#pragma unroll
      for (int dt = 0; dt < 4; ++dt) o[dt][e] *= alpha;
      // C-layout -> LDS so P can be re-loaded as an A fragment
#pragma unroll
      for (int nt = 0; nt < 4; ++nt)
        sP[wave][(rb + e) * 72 + nt * 16 + cl] = (bf16_t)sc[nt][e];
    }

    // O += P @ V : k = 64 keys -> 2 K-steps; 4 head-dim tiles
#pragma unroll
    for (int cc = 0; cc < 2; ++cc) {
      v16bf pf = load_frag(&sP[wave][cc * 32], 72, lane);
#pragma unroll
      for (int dt = 0; dt < 4; ++dt) {
        v16bf vf = load_frag(&sVt[(dt * 16) * 72 + cc * 32], 72, lane);
        o[dt] = wmma_bf16(pf, vf, o[dt]);
      }
    }
    __syncthreads();
  }

  // normalize and store ctx in [S][768] (heads concatenated)
#pragma unroll
  for (int dt = 0; dt < 4; ++dt) {
#pragma unroll
    for (int e = 0; e < 8; ++e) {
      const int row = q0 + rb + e;
      const int col = h * HDIM + dt * 16 + cl;
      ctx[(size_t)row * EMBED + col] = (bf16_t)(o[dt][e] / rsum[e]);
    }
  }
}

// ---------------------------------------------------------------------------
// out = LayerNorm(a + b) * gamma + beta ; optional bf16 copy for next GEMM
// One block per token row (768 elems, 256 threads x 3).
// ---------------------------------------------------------------------------
__global__ __launch_bounds__(256) void add_layernorm(
    const float* __restrict__ a, const float* __restrict__ b,
    const float* __restrict__ gamma, const float* __restrict__ beta,
    float* __restrict__ outF, bf16_t* __restrict__ outB) {
  const int row = blockIdx.x;
  const int tid = threadIdx.x;
  __shared__ float sred[16];
  __shared__ float sstat[2];

  float v[3];
  float s = 0.0f, s2 = 0.0f;
#pragma unroll
  for (int i = 0; i < 3; ++i) {
    const int cidx = tid + i * 256;
    const float t = a[(size_t)row * EMBED + cidx] + b[(size_t)row * EMBED + cidx];
    v[i] = t; s += t; s2 += t * t;
  }
#pragma unroll
  for (int m = 16; m >= 1; m >>= 1) { s += __shfl_xor(s, m); s2 += __shfl_xor(s2, m); }
  if ((tid & 31) == 0) { sred[tid >> 5] = s; sred[8 + (tid >> 5)] = s2; }
  __syncthreads();
  if (tid == 0) {
    float S = 0.0f, S2 = 0.0f;
    for (int w = 0; w < 8; ++w) { S += sred[w]; S2 += sred[8 + w]; }
    const float mu = S * (1.0f / EMBED);
    const float var = S2 * (1.0f / EMBED) - mu * mu;
    sstat[0] = mu;
    sstat[1] = rsqrtf(var + 1e-5f);
  }
  __syncthreads();
  const float mu = sstat[0], rstd = sstat[1];
#pragma unroll
  for (int i = 0; i < 3; ++i) {
    const int cidx = tid + i * 256;
    const float y = (v[i] - mu) * rstd * gamma[cidx] + beta[cidx];
    outF[(size_t)row * EMBED + cidx] = y;
    if (outB) outB[(size_t)row * EMBED + cidx] = (bf16_t)y;
  }
}

// ---------------------------------------------------------------------------
// Host orchestration
// ---------------------------------------------------------------------------
extern "C" void kernel_launch(void* const* d_in, const int* in_sizes, int n_in,
                              void* d_out, int out_size, void* d_ws, size_t ws_size,
                              hipStream_t stream) {
  (void)in_sizes; (void)n_in; (void)out_size; (void)ws_size;
  const int S = SEQ, D = EMBED, F = FFN_DIM, H = NHEADS;

  const float* x   = (const float*)d_in[0];
  const float* Wq  = (const float*)d_in[1];
  const float* bq  = (const float*)d_in[2];
  const float* Wk  = (const float*)d_in[3];
  const float* bk  = (const float*)d_in[4];
  const float* Wv  = (const float*)d_in[5];
  const float* bv  = (const float*)d_in[6];
  const float* Wo  = (const float*)d_in[7];
  const float* bo  = (const float*)d_in[8];
  const float* g1  = (const float*)d_in[9];
  const float* be1 = (const float*)d_in[10];
  const float* W1  = (const float*)d_in[11];
  const float* b1  = (const float*)d_in[12];
  const float* W2  = (const float*)d_in[13];
  const float* b2  = (const float*)d_in[14];
  const float* g2  = (const float*)d_in[15];
  const float* be2 = (const float*)d_in[16];

  char* ws = (char*)d_ws;
  size_t off = 0;
  auto alloc = [&](size_t bytes) -> void* {
    void* p = ws + off;
    off = (off + bytes + 255) & ~(size_t)255;
    return p;
  };
  bf16_t* Xb   = (bf16_t*)alloc((size_t)S * D * 2);
  bf16_t* Wqb  = (bf16_t*)alloc((size_t)D * D * 2);
  bf16_t* Wkb  = (bf16_t*)alloc((size_t)D * D * 2);
  bf16_t* Wvb  = (bf16_t*)alloc((size_t)D * D * 2);
  bf16_t* Wob  = (bf16_t*)alloc((size_t)D * D * 2);
  bf16_t* W1b  = (bf16_t*)alloc((size_t)D * F * 2);
  bf16_t* W2b  = (bf16_t*)alloc((size_t)F * D * 2);
  bf16_t* Qb   = (bf16_t*)alloc((size_t)S * D * 2);   // [H][S][64]
  bf16_t* Kb   = (bf16_t*)alloc((size_t)S * D * 2);
  bf16_t* Vb   = (bf16_t*)alloc((size_t)S * D * 2);
  bf16_t* Ctxb = (bf16_t*)alloc((size_t)S * D * 2);   // [S][768]
  float*  AttF = (float*) alloc((size_t)S * D * 4);
  float*  Hf   = (float*) alloc((size_t)S * D * 4);
  bf16_t* Hb   = (bf16_t*)alloc((size_t)S * D * 2);
  bf16_t* Gb   = (bf16_t*)alloc((size_t)S * F * 2);
  float*  FfnF = (float*) alloc((size_t)S * D * 4);

  auto cvt = [&](const float* src, bf16_t* dst, size_t n) {
    cvt_f32_bf16<<<(unsigned)((n + 255) / 256), 256, 0, stream>>>(src, dst, (int)n);
  };
  cvt(x,  Xb,  (size_t)S * D);
  cvt(Wq, Wqb, (size_t)D * D);
  cvt(Wk, Wkb, (size_t)D * D);
  cvt(Wv, Wvb, (size_t)D * D);
  cvt(Wo, Wob, (size_t)D * D);
  cvt(W1, W1b, (size_t)D * F);
  cvt(W2, W2b, (size_t)F * D);

  const dim3 blk(256);
  const size_t GEMM_LDS = 4 * 5120 * sizeof(bf16_t);   // 40 KB, double buffered
  // QKV projections, stored head-major for attention
  gemm_bf16_wmma<3><<<dim3(D / 128, S / 128), blk, GEMM_LDS, stream>>>(Xb, Wqb, bq, nullptr, Qb, S, D, D);
  gemm_bf16_wmma<3><<<dim3(D / 128, S / 128), blk, GEMM_LDS, stream>>>(Xb, Wkb, bk, nullptr, Kb, S, D, D);
  gemm_bf16_wmma<3><<<dim3(D / 128, S / 128), blk, GEMM_LDS, stream>>>(Xb, Wvb, bv, nullptr, Vb, S, D, D);
  // attention
  flash_attn_wmma<<<dim3(S / 128, H), blk, 0, stream>>>(Qb, Kb, Vb, Ctxb, S);
  // output projection (f32 for residual)
  gemm_bf16_wmma<0><<<dim3(D / 128, S / 128), blk, GEMM_LDS, stream>>>(Ctxb, Wob, bo, AttF, nullptr, S, D, D);
  // h = LN(x + attn_out), plus bf16 copy for FFN
  add_layernorm<<<S, 256, 0, stream>>>(x, AttF, g1, be1, Hf, Hb);
  // FFN
  gemm_bf16_wmma<2><<<dim3(F / 128, S / 128), blk, GEMM_LDS, stream>>>(Hb, W1b, b1, nullptr, Gb, S, F, D);
  gemm_bf16_wmma<0><<<dim3(D / 128, S / 128), blk, GEMM_LDS, stream>>>(Gb, W2b, b2, FfnF, nullptr, S, D, F);
  // out = LN(h + ffn)
  add_layernorm<<<S, 256, 0, stream>>>(Hf, FfnF, g2, be2, (float*)d_out, nullptr);
}
